// Attention_10514079940894
// MI455X (gfx1250) — matrix operand order repaired and checked by reading, not compiled, not run on hardware
//
#include <hip/hip_runtime.h>
#include <math.h>

// Problem constants (from reference)
#define BB 2
#define TT 2048
#define DD 2048
#define NHEADS 8
#define KHEADS 4
#define HH 256
#define WIN 1024
#define GQ (NHEADS / KHEADS)   // 2

typedef __attribute__((ext_vector_type(16))) __bf16 v16bf;
typedef __attribute__((ext_vector_type(8)))  float  v8f;
typedef __attribute__((ext_vector_type(4)))  unsigned int v4u;
typedef __attribute__((ext_vector_type(8)))  int v8i;
typedef __attribute__((ext_vector_type(4)))  int v4i;

union FragU {
  uint4 u[2];
  v16bf v;
  unsigned short s[16];
};

__device__ inline v16bf make_frag(const unsigned short* p0, const unsigned short* p1) {
  FragU f;
  f.u[0] = *(const uint4*)p0;
  f.u[1] = *(const uint4*)p1;
  return f.v;
}

__device__ inline v8f vzero() {
  v8f z;
#pragma unroll
  for (int i = 0; i < 8; ++i) z[i] = 0.0f;
  return z;
}

__device__ inline v8f wmma_bf16(v16bf a, v16bf b, v8f c) {
  // D = A(16x32 bf16) * B(32x16 bf16) + C(16x16 f32)
  return __builtin_amdgcn_wmma_f32_16x16x32_bf16(false, a, false, b, (short)0, c, false, false);
}

// ---- bf16 hi/lo split helpers (round-to-nearest-even) ----
__device__ inline unsigned short f2bf(float f) {
  unsigned int u = __float_as_uint(f);
  u += 0x7FFFu + ((u >> 16) & 1u);
  return (unsigned short)(u >> 16);
}
__device__ inline float bf2f(unsigned short h) {
  return __uint_as_float(((unsigned int)h) << 16);
}
__device__ inline void split_bf(float f, unsigned short& hi, unsigned short& lo) {
  hi = f2bf(f);
  lo = f2bf(f - bf2f(hi));
}

// =======================================================================
// TDM: 2-D tile load global -> LDS (ISA 08_async_tensor D# groups 0/1).
// Loads tile_rows x 32 bf16 elements, row pitch = row_stride elements,
// packed contiguously (row-major, 64B rows) at lds_byte_off.
// =======================================================================
__device__ inline void tdm_load_2d(const unsigned short* g, unsigned lds_byte_off,
                                   unsigned tile_rows, unsigned tensor_d0,
                                   unsigned tensor_d1, unsigned row_stride) {
  unsigned long long ga = (unsigned long long)g;
  v4u g0 = {
    1u,                                            // count=1, user descriptor
    lds_byte_off,                                  // lds_addr (bytes)
    (unsigned)(ga & 0xFFFFFFFFu),                  // global_addr[31:0]
    (unsigned)((ga >> 32) & 0x01FFFFFFu) | 0x80000000u  // global_addr[56:32] | type=2
  };
  v8i g1 = {
    (int)0x00010000u,                              // data_size=1 (2 bytes/elem)
    (int)((tensor_d0 & 0xFFFFu) << 16),            // tensor_dim0[15:0]
    (int)((tensor_d0 >> 16) | ((tensor_d1 & 0xFFFFu) << 16)),  // td0 hi | td1 lo
    (int)((tensor_d1 >> 16) | (32u << 16)),        // td1 hi | tile_dim0=32
    (int)tile_rows,                                // tile_dim1 (tile_dim2=0)
    (int)row_stride,                               // tensor_dim0_stride[31:0]
    0,                                             // stride hi | dim1_stride lo
    0
  };
  v4i gz = { 0, 0, 0, 0 };
#if __clang_major__ >= 23
  v8i gz8 = { 0, 0, 0, 0, 0, 0, 0, 0 };
  __builtin_amdgcn_tensor_load_to_lds(g0, g1, gz, gz, gz8, 0);
#else
  __builtin_amdgcn_tensor_load_to_lds(g0, g1, gz, gz, 0);
#endif
}

// =======================================================================
// Prep kernels: split f32 -> (hi,lo) bf16, with layout transforms so that
// every WMMA B-operand is stored [ncol][k] with k contiguous.
// =======================================================================
__global__ void k_split(const float* __restrict__ in, unsigned short* __restrict__ hi,
                        unsigned short* __restrict__ lo, size_t n) {
  size_t i = (size_t)blockIdx.x * blockDim.x + threadIdx.x;
  size_t st = (size_t)gridDim.x * blockDim.x;
  for (; i < n; i += st) {
    unsigned short h, l;
    split_bf(in[i], h, l);
    hi[i] = h; lo[i] = l;
  }
}

// in[hd][d][h]  ->  out[(hd*HH + h)][d]   (B-matrix transpose for projections)
__global__ void k_split_tr_dh(const float* __restrict__ in, unsigned short* __restrict__ hi,
                              unsigned short* __restrict__ lo, int heads) {
  size_t n = (size_t)heads * DD * HH;
  size_t i = (size_t)blockIdx.x * blockDim.x + threadIdx.x;
  size_t st = (size_t)gridDim.x * blockDim.x;
  for (; i < n; i += st) {
    int h = (int)(i % HH);
    size_t t = i / HH;
    int d = (int)(t % DD);
    int hd = (int)(t / DD);
    unsigned short hv, lv;
    split_bf(in[i], hv, lv);
    size_t o = ((size_t)hd * HH + h) * DD + d;
    hi[o] = hv; lo[o] = lv;
  }
}

// o_w[(n*HH+h)][d] -> owT[d][(n*HH+h)]
__global__ void k_split_ow(const float* __restrict__ in, unsigned short* __restrict__ hi,
                           unsigned short* __restrict__ lo) {
  size_t n = (size_t)NHEADS * HH * DD;
  size_t i = (size_t)blockIdx.x * blockDim.x + threadIdx.x;
  size_t st = (size_t)gridDim.x * blockDim.x;
  for (; i < n; i += st) {
    int d = (int)(i % DD);
    size_t nh = i / DD;
    unsigned short hv, lv;
    split_bf(in[i], hv, lv);
    size_t o = (size_t)d * (NHEADS * HH) + nh;
    hi[o] = hv; lo[o] = lv;
  }
}

// =======================================================================
// Kernel: fused QKV projection + RMSNorm + RoPE + q-scale.
// One wave computes a 16-row x 256-col (full head) strip; K = D = 2048.
// The shared 256x32 B-panel per K-step is staged into LDS by the Tensor
// Data Mover (each wave DMAs a 64-row chunk), double-buffered so the next
// panel's DMA overlaps the current panel's WMMAs.
// grid.y slot: 0..7 = Q head, 8..11 = K head, 12..15 = V head
// =======================================================================
__global__ __launch_bounds__(128) void k_qkv(
    const unsigned short* __restrict__ xhi,  const unsigned short* __restrict__ xlo,
    const unsigned short* __restrict__ qwThi, const unsigned short* __restrict__ qwTlo,
    const unsigned short* __restrict__ kwThi, const unsigned short* __restrict__ kwTlo,
    const unsigned short* __restrict__ vwThi, const unsigned short* __restrict__ vwTlo,
    const int* __restrict__ positions,
    const float* __restrict__ qns, const float* __restrict__ kns,
    unsigned short* __restrict__ qhi, unsigned short* __restrict__ qlo,
    unsigned short* __restrict__ khi, unsigned short* __restrict__ klo,
    unsigned short* __restrict__ vThi, unsigned short* __restrict__ vTlo) {
  __shared__ unsigned short ldsBh[2][256 * 32];
  __shared__ unsigned short ldsBl[2][256 * 32];

  const int lane = threadIdx.x & 31;
  const int wave = threadIdx.x >> 5;
  const int row0 = (blockIdx.x * 4 + wave) * 16;
  const int slot = blockIdx.y;
  const int ml = lane & 15;
  const int kh2 = lane >> 4;

  const unsigned short *wh, *wl;
  int head, type;
  if (slot < NHEADS)            { type = 0; head = slot;              wh = qwThi; wl = qwTlo; }
  else if (slot < NHEADS+KHEADS){ type = 1; head = slot - NHEADS;     wh = kwThi; wl = kwTlo; }
  else                          { type = 2; head = slot-NHEADS-KHEADS; wh = vwThi; wl = vwTlo; }

  const unsigned short* arh = xhi + (size_t)(row0 + ml) * DD;
  const unsigned short* arl = xlo + (size_t)(row0 + ml) * DD;
  const size_t hrow0 = (size_t)head * HH;

  const int w = __builtin_amdgcn_readfirstlane(wave);
  const unsigned ldsHi0 = (unsigned)(unsigned long long)(void*)&ldsBh[0][0];
  const unsigned ldsLo0 = (unsigned)(unsigned long long)(void*)&ldsBl[0][0];

  auto stage = [&](int kb, int buf) {
    const unsigned short* gh = wh + (hrow0 + (size_t)w * 64) * DD + kb;
    const unsigned short* gl = wl + (hrow0 + (size_t)w * 64) * DD + kb;
    unsigned off = (unsigned)buf * (256u * 32u * 2u) + (unsigned)w * (64u * 32u * 2u);
    tdm_load_2d(gh, ldsHi0 + off, 64, DD, 64, DD);
    tdm_load_2d(gl, ldsLo0 + off, 64, DD, 64, DD);
  };

  v8f acc[16];
  for (int c = 0; c < 16; ++c) acc[c] = vzero();

  stage(0, 0);  // prologue: first B-panel in flight

  int ib = 0;
  for (int kb = 0; kb < DD; kb += 32, ib ^= 1) {
    if (kb + 32 < DD) {
      stage(kb + 32, ib ^ 1);                 // issue next panel's DMA
      __builtin_amdgcn_s_wait_tensorcnt(2);   // older pair complete; newest 2 in flight
    } else {
      __builtin_amdgcn_s_wait_tensorcnt(0);
    }
    __syncthreads();                          // panel visible to all waves

    v16bf Ah = make_frag(arh + kb + kh2 * 16, arh + kb + kh2 * 16 + 8);
    v16bf Al = make_frag(arl + kb + kh2 * 16, arl + kb + kh2 * 16 + 8);
    __builtin_prefetch(arh + kb + 128, 0, 0);

    const unsigned short* baseH = &ldsBh[ib][0];
    const unsigned short* baseL = &ldsBl[ib][0];
#pragma unroll
    for (int c = 0; c < 16; ++c) {
      const unsigned short* bh = baseH + (c * 16 + ml) * 32 + kh2 * 16;
      const unsigned short* bl = baseL + (c * 16 + ml) * 32 + kh2 * 16;
      v16bf Bh = make_frag(bh, bh + 8);
      v16bf Bl = make_frag(bl, bl + 8);
      acc[c] = wmma_bf16(Al, Bh, acc[c]);   // lo*hi
      acc[c] = wmma_bf16(Ah, Bl, acc[c]);   // hi*lo
      acc[c] = wmma_bf16(Ah, Bh, acc[c]);   // hi*hi
    }
    __syncthreads();                          // all waves done before overwrite
  }

  if (type == 2) {
    // V: no norm/rope; store transposed [b][kh][h][t] (B-matrix for PV)
#pragma unroll
    for (int r = 0; r < 8; ++r) {
      int row = row0 + r + 8 * kh2;
      int b = row / TT, t = row % TT;
#pragma unroll
      for (int c = 0; c < 16; ++c) {
        unsigned short hv, lv;
        split_bf(acc[c][r], hv, lv);
        size_t o = (((size_t)(b * KHEADS + head) * HH + c * 16 + ml) * TT + t);
        vThi[o] = hv; vTlo[o] = lv;
      }
    }
    return;
  }

  const float* ns = (type == 0) ? qns : kns;
  float sc[16];
#pragma unroll
  for (int c = 0; c < 16; ++c) sc[c] = 1.0f + ns[c * 16 + ml];
  float invts[8];
#pragma unroll
  for (int c = 0; c < 8; ++c) {
    float j = (float)(c * 16 + ml);             // rope angle index 0..127
    invts[c] = __powf(10000.0f, -j * (1.0f / 128.0f));
  }

#pragma unroll
  for (int r = 0; r < 8; ++r) {
    int row = row0 + r + 8 * kh2;
    // RMS over the full head: partial over this lane's 16 columns, then
    // butterfly across the 16 lanes holding the same row.
    float ss = 0.0f;
#pragma unroll
    for (int c = 0; c < 16; ++c) ss += acc[c][r] * acc[c][r];
    ss += __shfl_xor(ss, 1); ss += __shfl_xor(ss, 2);
    ss += __shfl_xor(ss, 4); ss += __shfl_xor(ss, 8);
    float rs = rsqrtf(ss * (1.0f / HH) + 1e-6f);

    float y[16];
#pragma unroll
    for (int c = 0; c < 16; ++c) y[c] = acc[c][r] * rs * sc[c];

    float pos = (float)positions[row];
#pragma unroll
    for (int c = 0; c < 8; ++c) {             // pairs (h, h+128) live in (c, c+8)
      float s, co;
      __sincosf(pos * invts[c], &s, &co);
      float a = y[c], b2 = y[c + 8];
      y[c]     = a * co - b2 * s;
      y[c + 8] = b2 * co + a * s;
    }

    int b = row / TT, t = row % TT;
#pragma unroll
    for (int c = 0; c < 16; ++c) {
      float o = y[c];
      if (type == 0) o *= 0.0625f;            // H^-0.5
      unsigned short hv, lv;
      split_bf(o, hv, lv);
      if (type == 0) {
        size_t idx = ((size_t)row * NHEADS + head) * HH + c * 16 + ml;
        qhi[idx] = hv; qlo[idx] = lv;
      } else {
        size_t idx = (((size_t)(b * KHEADS + head) * TT + t) * HH + c * 16 + ml);
        khi[idx] = hv; klo[idx] = lv;
      }
    }
  }
}

// =======================================================================
// Flash attention: one wave per (b, head, 16-query tile); 32-key blocks,
// online softmax, P re-striped C-layout -> A-layout through LDS.
// =======================================================================
__global__ __launch_bounds__(128) void k_flash(
    const unsigned short* __restrict__ qhi, const unsigned short* __restrict__ qlo,
    const unsigned short* __restrict__ khi, const unsigned short* __restrict__ klo,
    const unsigned short* __restrict__ vThi, const unsigned short* __restrict__ vTlo,
    unsigned short* __restrict__ atthi, unsigned short* __restrict__ attlo) {
  __shared__ float pbuf[4][16 * 32];
  const int lane = threadIdx.x & 31;
  const int wave = threadIdx.x >> 5;
  const int gw = blockIdx.x * 4 + wave;
  const int QT = TT / 16;
  const int qt = gw % QT;
  const int hd = (gw / QT) % NHEADS;
  const int b  = gw / (QT * NHEADS);
  const int khd = hd / GQ;
  const int ml = lane & 15;
  const int kh2 = lane >> 4;

  // Q fragments resident in registers (8 K-steps, hi+lo)
  v16bf Qh[8], Ql[8];
  {
    const unsigned short* qh_ = qhi + (((size_t)(b * TT + qt * 16 + ml)) * NHEADS + hd) * HH;
    const unsigned short* ql_ = qlo + (((size_t)(b * TT + qt * 16 + ml)) * NHEADS + hd) * HH;
#pragma unroll
    for (int kk = 0; kk < 8; ++kk) {
      int kb = kk * 32;
      Qh[kk] = make_frag(qh_ + kb + kh2 * 16, qh_ + kb + kh2 * 16 + 8);
      Ql[kk] = make_frag(ql_ + kb + kh2 * 16, ql_ + kb + kh2 * 16 + 8);
    }
  }

  v8f O[16];
  for (int c = 0; c < 16; ++c) O[c] = vzero();
  float mrow[8], lrow[8];
#pragma unroll
  for (int r = 0; r < 8; ++r) { mrow[r] = -INFINITY; lrow[r] = 0.0f; }

  const int sHi = qt * 16 + 15;
  int s0 = qt * 16 - (WIN - 1);
  if (s0 < 0) s0 = 0;
  s0 &= ~31;

  float* pw = &pbuf[wave][0];

  for (int sb = s0; sb <= sHi; sb += 32) {
    v8f S0 = vzero(), S1 = vzero();
    const unsigned short* k0h = khi + (((size_t)(b * KHEADS + khd) * TT + sb + ml)) * HH;
    const unsigned short* k0l = klo + (((size_t)(b * KHEADS + khd) * TT + sb + ml)) * HH;
    const unsigned short* k1h = k0h + 16 * HH;
    const unsigned short* k1l = k0l + 16 * HH;
#pragma unroll
    for (int kk = 0; kk < 8; ++kk) {
      int off = kk * 32 + kh2 * 16;
      v16bf B0h = make_frag(k0h + off, k0h + off + 8);
      v16bf B0l = make_frag(k0l + off, k0l + off + 8);
      v16bf B1h = make_frag(k1h + off, k1h + off + 8);
      v16bf B1l = make_frag(k1l + off, k1l + off + 8);
      S0 = wmma_bf16(Ql[kk], B0h, S0);
      S0 = wmma_bf16(Qh[kk], B0l, S0);
      S0 = wmma_bf16(Qh[kk], B0h, S0);
      S1 = wmma_bf16(Ql[kk], B1h, S1);
      S1 = wmma_bf16(Qh[kk], B1l, S1);
      S1 = wmma_bf16(Qh[kk], B1h, S1);
    }

    // windowed-causal mask + online softmax (C layout: lane holds key col)
#pragma unroll
    for (int r = 0; r < 8; ++r) {
      int m_ = r + 8 * kh2;
      int qp = qt * 16 + m_;
      int sA = sb + ml, sB = sA + 16;
      float v0 = ((sA <= qp) && (sA > qp - WIN)) ? S0[r] : -INFINITY;
      float v1 = ((sB <= qp) && (sB > qp - WIN)) ? S1[r] : -INFINITY;
      float rmax = fmaxf(v0, v1);
      rmax = fmaxf(rmax, __shfl_xor(rmax, 1));
      rmax = fmaxf(rmax, __shfl_xor(rmax, 2));
      rmax = fmaxf(rmax, __shfl_xor(rmax, 4));
      rmax = fmaxf(rmax, __shfl_xor(rmax, 8));
      float mnew = fmaxf(mrow[r], rmax);
      float alpha = (mrow[r] == -INFINITY) ? 0.0f : __expf(mrow[r] - mnew);
      float p0 = (v0 == -INFINITY) ? 0.0f : __expf(v0 - mnew);
      float p1 = (v1 == -INFINITY) ? 0.0f : __expf(v1 - mnew);
      float rs = p0 + p1;
      rs += __shfl_xor(rs, 1); rs += __shfl_xor(rs, 2);
      rs += __shfl_xor(rs, 4); rs += __shfl_xor(rs, 8);
      lrow[r] = lrow[r] * alpha + rs;
      mrow[r] = mnew;
#pragma unroll
      for (int c = 0; c < 16; ++c) O[c][r] = O[c][r] * alpha;
      pw[m_ * 32 + ml] = p0;
      pw[m_ * 32 + ml + 16] = p1;
    }

    // restripe P (C layout in LDS) into A-layout bf16 hi/lo fragments
    FragU fh, fl;
    const float* pr = &pw[ml * 32];
#pragma unroll
    for (int j = 0; j < 8; ++j) {
      split_bf(pr[kh2 * 16 + j],     fh.s[j],     fl.s[j]);
      split_bf(pr[kh2 * 16 + 8 + j], fh.s[8 + j], fl.s[8 + j]);
    }
    v16bf Ph = fh.v, Pl = fl.v;

    // PV: O(16x256) += P(16x32) @ V(32x256)
#pragma unroll
    for (int c = 0; c < 16; ++c) {
      const unsigned short* vh = vThi + (((size_t)(b * KHEADS + khd) * HH + c * 16 + ml)) * TT + sb + kh2 * 16;
      const unsigned short* vl = vTlo + (((size_t)(b * KHEADS + khd) * HH + c * 16 + ml)) * TT + sb + kh2 * 16;
      v16bf Bh = make_frag(vh, vh + 8);
      v16bf Bl = make_frag(vl, vl + 8);
      O[c] = wmma_bf16(Pl, Bh, O[c]);
      O[c] = wmma_bf16(Ph, Bl, O[c]);
      O[c] = wmma_bf16(Ph, Bh, O[c]);
    }
  }

  // normalize + store att (hi/lo bf16, [b,t,n,h] so (n,h) is contiguous K for o-proj)
#pragma unroll
  for (int r = 0; r < 8; ++r) {
    int m_ = r + 8 * kh2;
    float inv = 1.0f / lrow[r];
    size_t base = (((size_t)(b * TT + qt * 16 + m_)) * NHEADS + hd) * HH;
#pragma unroll
    for (int c = 0; c < 16; ++c) {
      unsigned short hv, lv;
      split_bf(O[c][r] * inv, hv, lv);
      atthi[base + c * 16 + ml] = hv;
      attlo[base + c * 16 + ml] = lv;
    }
  }
}

// =======================================================================
// Output projection: out[bt][d] = att[bt][(n,h)] @ o_w[(n,h)][d]
// Same TDM-staged, double-buffered B-panel scheme as k_qkv.
// =======================================================================
__global__ __launch_bounds__(128) void k_oproj(
    const unsigned short* __restrict__ ahi, const unsigned short* __restrict__ alo,
    const unsigned short* __restrict__ owThi, const unsigned short* __restrict__ owTlo,
    float* __restrict__ out) {
  __shared__ unsigned short ldsBh[2][256 * 32];
  __shared__ unsigned short ldsBl[2][256 * 32];

  const int lane = threadIdx.x & 31;
  const int wave = threadIdx.x >> 5;
  const int row0 = (blockIdx.x * 4 + wave) * 16;
  const int n0 = blockIdx.y * 256;
  const int ml = lane & 15;
  const int kh2 = lane >> 4;
  const int KD = NHEADS * HH;   // 2048

  const unsigned short* arh = ahi + (size_t)(row0 + ml) * KD;
  const unsigned short* arl = alo + (size_t)(row0 + ml) * KD;

  const int w = __builtin_amdgcn_readfirstlane(wave);
  const unsigned ldsHi0 = (unsigned)(unsigned long long)(void*)&ldsBh[0][0];
  const unsigned ldsLo0 = (unsigned)(unsigned long long)(void*)&ldsBl[0][0];

  auto stage = [&](int kb, int buf) {
    const unsigned short* gh = owThi + (size_t)(n0 + w * 64) * KD + kb;
    const unsigned short* gl = owTlo + (size_t)(n0 + w * 64) * KD + kb;
    unsigned off = (unsigned)buf * (256u * 32u * 2u) + (unsigned)w * (64u * 32u * 2u);
    tdm_load_2d(gh, ldsHi0 + off, 64, KD, 64, KD);
    tdm_load_2d(gl, ldsLo0 + off, 64, KD, 64, KD);
  };

  v8f acc[16];
  for (int c = 0; c < 16; ++c) acc[c] = vzero();

  stage(0, 0);

  int ib = 0;
  for (int kb = 0; kb < KD; kb += 32, ib ^= 1) {
    if (kb + 32 < KD) {
      stage(kb + 32, ib ^ 1);
      __builtin_amdgcn_s_wait_tensorcnt(2);
    } else {
      __builtin_amdgcn_s_wait_tensorcnt(0);
    }
    __syncthreads();

    v16bf Ah = make_frag(arh + kb + kh2 * 16, arh + kb + kh2 * 16 + 8);
    v16bf Al = make_frag(arl + kb + kh2 * 16, arl + kb + kh2 * 16 + 8);
    __builtin_prefetch(arh + kb + 128, 0, 0);

    const unsigned short* baseH = &ldsBh[ib][0];
    const unsigned short* baseL = &ldsBl[ib][0];
#pragma unroll
    for (int c = 0; c < 16; ++c) {
      const unsigned short* bh = baseH + (c * 16 + ml) * 32 + kh2 * 16;
      const unsigned short* bl = baseL + (c * 16 + ml) * 32 + kh2 * 16;
      v16bf Bh = make_frag(bh, bh + 8);
      v16bf Bl = make_frag(bl, bl + 8);
      acc[c] = wmma_bf16(Al, Bh, acc[c]);
      acc[c] = wmma_bf16(Ah, Bl, acc[c]);
      acc[c] = wmma_bf16(Ah, Bh, acc[c]);
    }
    __syncthreads();
  }

#pragma unroll
  for (int r = 0; r < 8; ++r) {
    size_t row = (size_t)(row0 + r + 8 * kh2);
#pragma unroll
    for (int c = 0; c < 16; ++c) out[row * DD + n0 + c * 16 + ml] = acc[c][r];
  }
}

// =======================================================================
extern "C" void kernel_launch(void* const* d_in, const int* in_sizes, int n_in,
                              void* d_out, int out_size, void* d_ws, size_t ws_size,
                              hipStream_t stream) {
  const float* x         = (const float*)d_in[0];
  const int*   positions = (const int*)d_in[1];
  const float* q_w       = (const float*)d_in[2];
  const float* k_w       = (const float*)d_in[3];
  const float* v_w       = (const float*)d_in[4];
  const float* o_w       = (const float*)d_in[5];
  const float* qns       = (const float*)d_in[6];
  const float* kns       = (const float*)d_in[7];
  float* out = (float*)d_out;

  char* ws = (char*)d_ws;
  size_t off = 0;
  auto carve = [&](size_t elems) -> unsigned short* {
    unsigned short* p = (unsigned short*)(ws + off);
    off = (off + elems * sizeof(unsigned short) + 255) & ~(size_t)255;
    return p;
  };

  const size_t BT = (size_t)BB * TT;
  unsigned short* xhi   = carve(BT * DD);
  unsigned short* xlo   = carve(BT * DD);
  unsigned short* qwThi = carve((size_t)NHEADS * HH * DD);
  unsigned short* qwTlo = carve((size_t)NHEADS * HH * DD);
  unsigned short* kwThi = carve((size_t)KHEADS * HH * DD);
  unsigned short* kwTlo = carve((size_t)KHEADS * HH * DD);
  unsigned short* vwThi = carve((size_t)KHEADS * HH * DD);
  unsigned short* vwTlo = carve((size_t)KHEADS * HH * DD);
  unsigned short* owThi = carve((size_t)DD * NHEADS * HH);
  unsigned short* owTlo = carve((size_t)DD * NHEADS * HH);
  unsigned short* qhi   = carve(BT * NHEADS * HH);
  unsigned short* qlo   = carve(BT * NHEADS * HH);
  unsigned short* khi   = carve(BT * KHEADS * HH);
  unsigned short* klo   = carve(BT * KHEADS * HH);
  unsigned short* vThi  = carve(BT * KHEADS * HH);
  unsigned short* vTlo  = carve(BT * KHEADS * HH);
  unsigned short* atthi = carve(BT * NHEADS * HH);
  unsigned short* attlo = carve(BT * NHEADS * HH);

  // 1) split/transpose prep
  k_split<<<4096, 256, 0, stream>>>(x, xhi, xlo, BT * DD);
  k_split_tr_dh<<<4096, 256, 0, stream>>>(q_w, qwThi, qwTlo, NHEADS);
  k_split_tr_dh<<<2048, 256, 0, stream>>>(k_w, kwThi, kwTlo, KHEADS);
  k_split_tr_dh<<<2048, 256, 0, stream>>>(v_w, vwThi, vwTlo, KHEADS);
  k_split_ow<<<4096, 256, 0, stream>>>(o_w, owThi, owTlo);

  // 2) fused QKV projection + RMSNorm + RoPE (TDM-staged B panels)
  k_qkv<<<dim3((unsigned)(BT / 64), NHEADS + 2 * KHEADS), 128, 0, stream>>>(
      xhi, xlo, qwThi, qwTlo, kwThi, kwTlo, vwThi, vwTlo,
      positions, qns, kns, qhi, qlo, khi, klo, vThi, vTlo);

  // 3) windowed flash attention
  k_flash<<<(BB * NHEADS * (TT / 16)) / 4, 128, 0, stream>>>(
      qhi, qlo, khi, klo, vThi, vTlo, atthi, attlo);

  // 4) output projection (TDM-staged B panels)
  k_oproj<<<dim3((unsigned)(BT / 64), DD / 256), 128, 0, stream>>>(
      atthi, attlo, owThi, owTlo, out);
}
